// GCN_22703197127026
// MI455X (gfx1250) — matrix operand order repaired and checked by reading, not compiled, not run on hardware
//
#include <hip/hip_runtime.h>

// ---------------------------------------------------------------------------
// GCN forward for MI455X (gfx1250, wave32, WMMA).
//   H0 = X@W0            (bf16 WMMA, f32 accum)
//   H1 = relu(spmm(H0))  (f32 atomic scatter-add; relu fused into GEMM2 load)
//   H2 = H1@W1           (bf16 WMMA, f32 accum)
//   Z  = spmm(H2)        (f32 atomic scatter-add)
//   out= softmax(Z)      (wave-per-row)
// ---------------------------------------------------------------------------

typedef __attribute__((ext_vector_type(16))) __bf16          v16bf;
typedef __attribute__((ext_vector_type(16))) unsigned short  v16us;
typedef __attribute__((ext_vector_type(8)))  unsigned short  v8us;
typedef __attribute__((ext_vector_type(8)))  float           v8f;

#define F0 512   // input feature dim
#define F1 256   // hidden dim
#define F2 64    // output classes
#define LDSPITCH 40  // ushort pitch for LDS A tiles (80B rows: conflict-free)

union Frag16 { v16us u; v16bf b; };

__device__ __forceinline__ unsigned short f32_to_bf16(float f) {
  unsigned int u = __float_as_uint(f);
  u += 0x7FFFu + ((u >> 16) & 1u);      // round-to-nearest-even
  return (unsigned short)(u >> 16);
}

// --- weight pre-conversion: W0[512,256]->W0T bf16 [256][512],
//                            W1[256, 64]->W1T bf16 [ 64][256] ---------------
__global__ __launch_bounds__(256) void convert_weights(
    const float* __restrict__ W0, const float* __restrict__ W1,
    unsigned short* __restrict__ W0T, unsigned short* __restrict__ W1T) {
  int i = blockIdx.x * 256 + threadIdx.x;
  if (i < F0 * F1) {
    int k = i / F1, c = i % F1;
    W0T[c * F0 + k] = f32_to_bf16(W0[i]);
  } else if (i < F0 * F1 + F1 * F2) {
    int j = i - F0 * F1;
    int k = j / F2, c = j % F2;
    W1T[c * F1 + k] = f32_to_bf16(W1[j]);
  }
}

// --- GEMM1: H0[N,256] = X[N,512] @ W0 ; 16-row strip per block -------------
__global__ __launch_bounds__(256) void gemm_xw0(
    const float* __restrict__ X, const unsigned short* __restrict__ W0T,
    float* __restrict__ H0, int N) {
  __shared__ __align__(32) unsigned short As[16 * LDSPITCH];
  const int tid     = threadIdx.x;
  const int wave    = tid >> 5;
  const int lane    = tid & 31;
  const int halfsel = lane >> 4;            // lanes 16-31 hold high K-halves
  const int l15     = lane & 15;
  const int rowBase = blockIdx.x * 16;

  v8f acc0 = {};
  v8f acc1 = {};
  const int col0 = wave * 32 + l15;         // each wave owns two 16-col tiles
  const int col1 = col0 + 16;

  // LDS staging indices: 512 elems (16 rows x 32 k), 2 per thread, float2
  const int e  = tid * 2;
  const int sr = e >> 5;
  const int sk = e & 31;
  int srow = rowBase + sr; if (srow >= N) srow = N - 1;
  const float* xrow = X + (size_t)srow * F0;

  for (int kb = 0; kb < F0; kb += 32) {
    float2 xv = *(const float2*)(xrow + kb + sk);
    unsigned int packed = (unsigned int)f32_to_bf16(xv.x) |
                          ((unsigned int)f32_to_bf16(xv.y) << 16);
    *(unsigned int*)&As[sr * LDSPITCH + sk] = packed;
    __syncthreads();

    // A fragment: row = l15; lanes<16 -> K {0..15}, lanes>=16 -> {16..31}
    Frag16 af;
    {
      const int k0 = halfsel ? 16 : 0;
      v8us lo = *(const v8us*)&As[l15 * LDSPITCH + k0];        // ds_load_b128
      v8us hi = *(const v8us*)&As[l15 * LDSPITCH + k0 + 8];
#pragma unroll
      for (int i = 0; i < 8; ++i) { af.u[i] = lo[i]; af.u[8 + i] = hi[i]; }
    }
    // B fragments: 16 contiguous bf16 K-values for this lane's column
    const int kh = kb + (halfsel ? 16 : 0);
    Frag16 b0, b1;
    b0.u = *(const v16us*)&W0T[(size_t)col0 * F0 + kh];        // 32B load
    b1.u = *(const v16us*)&W0T[(size_t)col1 * F0 + kh];

    acc0 = __builtin_amdgcn_wmma_f32_16x16x32_bf16(false, af.b, false, b0.b,
                                                   (short)0, acc0, false, false);
    acc1 = __builtin_amdgcn_wmma_f32_16x16x32_bf16(false, af.b, false, b1.b,
                                                   (short)0, acc1, false, false);
    __syncthreads();
  }

#pragma unroll
  for (int r = 0; r < 8; ++r) {             // C layout: M = r + 8*halfsel, N = l15
    int row = rowBase + r + halfsel * 8;
    if (row < N) {
      H0[(size_t)row * F1 + col0] = acc0[r];
      H0[(size_t)row * F1 + col1] = acc1[r];
    }
  }
}

// --- GEMM2: H2[N,64] = relu(H1)[N,256] @ W1 ; 32x64 tile per block ---------
__global__ __launch_bounds__(256) void gemm_h1w1(
    const float* __restrict__ H1, const unsigned short* __restrict__ W1T,
    float* __restrict__ H2, int N) {
  __shared__ __align__(32) unsigned short As[32 * LDSPITCH];
  const int tid     = threadIdx.x;
  const int wave    = tid >> 5;
  const int lane    = tid & 31;
  const int halfsel = lane >> 4;
  const int l15     = lane & 15;
  const int rowTile = wave >> 2;            // 0..1
  const int colTile = wave & 3;             // 0..3
  const int rowBase = blockIdx.x * 32;

  v8f acc = {};
  const int col = colTile * 16 + l15;

  // staging: 1024 elems (32 rows x 32 k), 4 per thread, float4 + fused relu
  const int e  = tid * 4;
  const int sr = e >> 5;
  const int sk = e & 31;
  int srow = rowBase + sr; if (srow >= N) srow = N - 1;
  const float* hrow = H1 + (size_t)srow * F1;

  for (int kb = 0; kb < F1; kb += 32) {
    float4 hv = *(const float4*)(hrow + kb + sk);
    hv.x = fmaxf(hv.x, 0.f); hv.y = fmaxf(hv.y, 0.f);
    hv.z = fmaxf(hv.z, 0.f); hv.w = fmaxf(hv.w, 0.f);
    unsigned int p0 = (unsigned int)f32_to_bf16(hv.x) |
                      ((unsigned int)f32_to_bf16(hv.y) << 16);
    unsigned int p1 = (unsigned int)f32_to_bf16(hv.z) |
                      ((unsigned int)f32_to_bf16(hv.w) << 16);
    unsigned int* dst = (unsigned int*)&As[sr * LDSPITCH + sk];
    dst[0] = p0; dst[1] = p1;
    __syncthreads();

    Frag16 af;
    {
      const int arow = rowTile * 16 + l15;
      const int k0 = halfsel ? 16 : 0;
      v8us lo = *(const v8us*)&As[arow * LDSPITCH + k0];
      v8us hi = *(const v8us*)&As[arow * LDSPITCH + k0 + 8];
#pragma unroll
      for (int i = 0; i < 8; ++i) { af.u[i] = lo[i]; af.u[8 + i] = hi[i]; }
    }
    Frag16 bf_;
    bf_.u = *(const v16us*)&W1T[(size_t)col * F1 + kb + (halfsel ? 16 : 0)];

    acc = __builtin_amdgcn_wmma_f32_16x16x32_bf16(false, af.b, false, bf_.b,
                                                  (short)0, acc, false, false);
    __syncthreads();
  }

#pragma unroll
  for (int r = 0; r < 8; ++r) {
    int row = rowBase + rowTile * 16 + r + halfsel * 8;
    if (row < N) H2[(size_t)row * F2 + col] = acc[r];
  }
}

// --- zero-init (float4 grid-stride) ----------------------------------------
__global__ __launch_bounds__(256) void zero_f32v4(float4* __restrict__ p, size_t n4) {
  size_t i = (size_t)blockIdx.x * 256 + threadIdx.x;
  size_t stride = (size_t)gridDim.x * 256;
  float4 z; z.x = z.y = z.z = z.w = 0.f;
  for (; i < n4; i += stride) p[i] = z;
}

// --- SpMM scatter-add, F=256: 64 lanes x float4 per edge, 16 edges/block ---
__global__ __launch_bounds__(256) void spmm_f256(
    const int* __restrict__ rows, const int* __restrict__ cols,
    const float* __restrict__ vals, const float* __restrict__ Y,
    float* __restrict__ out, int E) {
  const int lane = threadIdx.x & 63;
  const int slot = threadIdx.x >> 6;
  const int e0 = blockIdx.x * 16 + slot;
#pragma unroll
  for (int i = 0; i < 4; ++i) {
    int e = e0 + i * 4;
    if (e < E) {
      int r = rows[e], c = cols[e];
      float v = vals[e];
      float4 y = *(const float4*)(Y + (size_t)c * F1 + lane * 4);
      float* dst = out + (size_t)r * F1 + lane * 4;
      atomicAdd(dst + 0, v * y.x);
      atomicAdd(dst + 1, v * y.y);
      atomicAdd(dst + 2, v * y.z);
      atomicAdd(dst + 3, v * y.w);
    }
  }
}

// --- SpMM scatter-add, F=64: 16 lanes x float4 per edge, 64 edges/block ----
__global__ __launch_bounds__(256) void spmm_f64(
    const int* __restrict__ rows, const int* __restrict__ cols,
    const float* __restrict__ vals, const float* __restrict__ Y,
    float* __restrict__ out, int E) {
  const int lane = threadIdx.x & 15;
  const int slot = threadIdx.x >> 4;
  const int e0 = blockIdx.x * 64 + slot;
#pragma unroll
  for (int i = 0; i < 4; ++i) {
    int e = e0 + i * 16;
    if (e < E) {
      int r = rows[e], c = cols[e];
      float v = vals[e];
      float4 y = *(const float4*)(Y + (size_t)c * F2 + lane * 4);
      float* dst = out + (size_t)r * F2 + lane * 4;
      atomicAdd(dst + 0, v * y.x);
      atomicAdd(dst + 1, v * y.y);
      atomicAdd(dst + 2, v * y.z);
      atomicAdd(dst + 3, v * y.w);
    }
  }
}

// --- row softmax over 64 classes: one wave (32 lanes x 2 vals) per row -----
__global__ __launch_bounds__(256) void softmax64(
    const float* __restrict__ Z, float* __restrict__ out, int N) {
  const int wave = threadIdx.x >> 5;
  const int lane = threadIdx.x & 31;
  const int row = blockIdx.x * 8 + wave;
  if (row >= N) return;
  const float* z = Z + (size_t)row * F2;
  float a = z[lane], b = z[lane + 32];
  float m = fmaxf(a, b);
#pragma unroll
  for (int off = 16; off >= 1; off >>= 1) m = fmaxf(m, __shfl_xor(m, off));
  float ea = __expf(a - m), eb = __expf(b - m);
  float s = ea + eb;
#pragma unroll
  for (int off = 16; off >= 1; off >>= 1) s += __shfl_xor(s, off);
  float inv = 1.0f / s;
  float* o = out + (size_t)row * F2;
  o[lane] = ea * inv;
  o[lane + 32] = eb * inv;
}

// ---------------------------------------------------------------------------
extern "C" void kernel_launch(void* const* d_in, const int* in_sizes, int n_in,
                              void* d_out, int out_size, void* d_ws, size_t ws_size,
                              hipStream_t stream) {
  const float* X    = (const float*)d_in[0];
  const int*   erow = (const int*)d_in[1];
  const int*   ecol = (const int*)d_in[2];
  const float* eval = (const float*)d_in[3];
  const float* W0   = (const float*)d_in[4];
  const float* W1   = (const float*)d_in[5];
  const int N = in_sizes[0] / F0;   // 100000
  const int E = in_sizes[1];        // 3200000

  // workspace layout (bufA is reused: H0 dies after spmm1, then H2 + Z live there)
  char* ws = (char*)d_ws;
  unsigned short* W0T = (unsigned short*)ws;                       // 262144 B
  unsigned short* W1T = (unsigned short*)(ws + 262144);            //  32768 B
  float* bufA = (float*)(ws + 294912);                             // N*256 f32
  float* H0 = bufA;
  float* H1 = (float*)(ws + 294912 + (size_t)N * F1 * sizeof(float)); // N*256 f32
  float* H2 = bufA;                                                // N*64 f32
  float* Z  = bufA + (size_t)N * F2;                               // N*64 f32

  const int wconv_grid = (F0 * F1 + F1 * F2 + 255) / 256;
  convert_weights<<<wconv_grid, 256, 0, stream>>>(W0, W1, W0T, W1T);

  gemm_xw0<<<(N + 15) / 16, 256, 0, stream>>>(X, W0T, H0, N);

  zero_f32v4<<<2048, 256, 0, stream>>>((float4*)H1, (size_t)N * F1 / 4);
  spmm_f256<<<(E + 15) / 16, 256, 0, stream>>>(erow, ecol, eval, H0, H1, E);

  gemm_h1w1<<<(N + 31) / 32, 256, 0, stream>>>(H1, W1T, H2, N);

  zero_f32v4<<<1024, 256, 0, stream>>>((float4*)Z, (size_t)N * F2 / 4);
  spmm_f64<<<(E + 63) / 64, 256, 0, stream>>>(erow, ecol, eval, H2, Z, E);

  softmax64<<<(N + 7) / 8, 256, 0, stream>>>(Z, (float*)d_out, N);
}
